// Head_46359876993718
// MI455X (gfx1250) — matrix operand order repaired and checked by reading, not compiled, not run on hardware
//
#include <hip/hip_runtime.h>

// Causal single-head attention for MI455X (gfx1250, wave32).
// Stage 0: one-time W -> W^T bf16 transpose (makes GEMM B-fragments contiguous)
// Stage 1: QKV projection via v_wmma_f32_16x16x32_bf16 (C^-0.5 folded into Q)
// Stage 2: flash attention computing S^T / O^T so softmax reductions are
//          in-lane + one shfl_xor(16); K tile staged into LDS by the
//          Tensor Data Mover (tensor_load_to_lds + s_wait_tensorcnt).

#define BB 4
#define TT 4096
#define CC 1024
#define HH 64

typedef __attribute__((ext_vector_type(16))) __bf16 v16bf;
typedef __attribute__((ext_vector_type(8)))  float  v8f;

#if __has_builtin(__builtin_amdgcn_tensor_load_to_lds) && \
    __has_builtin(__builtin_amdgcn_s_wait_tensorcnt)
#define HAVE_TDM 1
typedef __attribute__((ext_vector_type(4))) unsigned tdm_u4;
typedef __attribute__((ext_vector_type(8))) int      tdm_i8;
typedef __attribute__((ext_vector_type(4))) int      tdm_i4;
#else
#define HAVE_TDM 0
#endif

typedef __attribute__((address_space(3))) char lds_char_t;
__device__ inline unsigned lds_offset(const void* p) {
  // ptrtoint of an addrspace(3) pointer == raw LDS byte offset
  return (unsigned)(unsigned long long)(lds_char_t*)p;
}

__device__ inline __bf16 f2bf(float f) { return (__bf16)f; }  // native cvt

__device__ inline v8f v8zero() {
  v8f z = {0.f,0.f,0.f,0.f,0.f,0.f,0.f,0.f};
  return z;
}
__device__ inline v8f wmma_bf16(v16bf a, v16bf b, v8f c) {
  return __builtin_amdgcn_wmma_f32_16x16x32_bf16(false, a, false, b,
                                                 (short)0, c, false, false);
}

union Frag16 {
  v16bf v;
  uint4 q[2];
  __bf16 e[16];
};

// Load a 16-element bf16 fragment as two contiguous 8-element (16B) chunks
// at row+base and row+base+16 (matches A-fragment per-lane chunking).
__device__ inline v16bf lds_frag(const __bf16* row, int base) {
  Frag16 f;
  f.q[0] = *(const uint4*)(row + base);
  f.q[1] = *(const uint4*)(row + base + 16);
  return f.v;
}

// ---------------- Stage 0: W -> W^T (bf16) ---------------------------------
// grid = (64, 3), 256 threads. Block (col, m) writes row `col` of W_m^T.
__global__ void __launch_bounds__(256) wtrans(
    const float* __restrict__ Wq, const float* __restrict__ Wk,
    const float* __restrict__ Wv,
    __bf16* __restrict__ WqT, __bf16* __restrict__ WkT,
    __bf16* __restrict__ WvT) {
  const int col = blockIdx.x;                 // 0..63 (output column of W)
  const float* src;
  __bf16* dst;
  if (blockIdx.y == 0)      { src = Wq; dst = WqT; }
  else if (blockIdx.y == 1) { src = Wk; dst = WkT; }
  else                      { src = Wv; dst = WvT; }
  for (int k = threadIdx.x; k < CC; k += 256)
    dst[(size_t)col * CC + k] = f2bf(src[(size_t)k * HH + col]);
}

// ---------------- Stage 1: Q/K/V projection --------------------------------
// grid = (B*T)/16 blocks, 128 threads (4 waves). Wave w computes output
// columns [16w,16w+16) of all three projections for this block's 16 rows.
// B-fragments are single contiguous v16bf loads from W^T.
__global__ void __launch_bounds__(128) proj_qkv(
    const float* __restrict__ x,
    const __bf16* __restrict__ WqT, const __bf16* __restrict__ WkT,
    const __bf16* __restrict__ WvT,
    __bf16* __restrict__ Qw, __bf16* __restrict__ Kw, __bf16* __restrict__ Vw) {
  const int tid  = threadIdx.x;
  const int wave = tid >> 5;
  const int lane = tid & 31;
  const int hl   = lane >> 4;      // half-lane (K split per WMMA layout)
  const int ln   = lane & 15;
  const int rowBase = blockIdx.x * 16;
  const int n0   = wave * 16;
  const int col  = n0 + ln;

  v8f aq = v8zero(), ak = v8zero(), av = v8zero();
  const float*  xrow = x + (size_t)(rowBase + ln) * CC;
  const __bf16* wqr  = WqT + (size_t)col * CC;
  const __bf16* wkr  = WkT + (size_t)col * CC;
  const __bf16* wvr  = WvT + (size_t)col * CC;

  for (int kc = 0; kc < CC; kc += 32) {
    // keep the HBM-heavy x stream ahead of the math
    __builtin_prefetch(xrow + kc + 128, 0, 1);

    // A fragment: row m = ln; lanes<16 hold K = kc+[0..15],
    // lanes>=16 hold K = kc+[16..31].
    const float* p1 = xrow + kc + 16 * hl;
    const float* p2 = xrow + kc + 16 * hl + 8;
    float4 f0 = ((const float4*)p1)[0], f1 = ((const float4*)p1)[1];
    float4 g0 = ((const float4*)p2)[0], g1 = ((const float4*)p2)[1];
    float av16[16] = {f0.x,f0.y,f0.z,f0.w, f1.x,f1.y,f1.z,f1.w,
                      g0.x,g0.y,g0.z,g0.w, g1.x,g1.y,g1.z,g1.w};
    Frag16 a;
    #pragma unroll
    for (int i = 0; i < 16; ++i) a.e[i] = f2bf(av16[i]);

    // B fragments: lane holds 16 consecutive K-dims of its column, i.e. one
    // 32B chunk of a W^T row. lanes<16: kc+[0..15], lanes>=16: kc+[16..31].
    const int kd = kc + 16 * hl;
    v16bf bq = *(const v16bf*)(wqr + kd);
    v16bf bk = *(const v16bf*)(wkr + kd);
    v16bf bv = *(const v16bf*)(wvr + kd);

    aq = wmma_bf16(a.v, bq, aq);
    ak = wmma_bf16(a.v, bk, ak);
    av = wmma_bf16(a.v, bv, av);
  }

  // D layout: lane n holds column n; VGPR r holds row r (lanes<16) / r+8.
  #pragma unroll
  for (int r = 0; r < 8; ++r) {
    int row = rowBase + r + 8 * hl;
    size_t idx = (size_t)row * HH + col;
    Qw[idx] = f2bf(aq[r] * 0.03125f);   // fold C^-0.5 = 1/32 into Q
    Kw[idx] = f2bf(ak[r]);
    Vw[idx] = f2bf(av[r]);
  }
}

// ---------------- Stage 2: flash attention ---------------------------------
// grid = (T/64, B), 128 threads (4 waves). Wave w owns 16 query rows.
// Per 32-key tile: S^T = K_tile(32x64) @ Q^T(64x16)  (4 WMMAs)
//                  O^T += V^T(64x32) @ P^T(32x16)    (4 WMMAs)
__global__ void __launch_bounds__(128) attn_fwd(
    const __bf16* __restrict__ Qw, const __bf16* __restrict__ Kw,
    const __bf16* __restrict__ Vw, float* __restrict__ out) {
  __shared__ __bf16 kT[32][64];   // key-major K tile (TDM destination)
  __shared__ __bf16 vT[64][32];   // head-major (transposed) V tile

  const int tid  = threadIdx.x;
  const int wave = tid >> 5;
  const int lane = tid & 31;
  const int hl   = lane >> 4;
  const int ln   = lane & 15;
  const int b    = blockIdx.y;
  const int qblk = blockIdx.x * 64;
  const int qg   = qblk + wave * 16 + ln;   // this lane's query (S^T column)

#if HAVE_TDM
  const unsigned ldsK = lds_offset(&kT[0][0]);
#endif

  // Q^T B-fragments, register-resident for the whole KV loop.
  const __bf16* qrow = Qw + ((size_t)b * TT + qg) * HH;
  v16bf bq0 = lds_frag(qrow, 8 * hl);        // head dims  0..31
  v16bf bq1 = lds_frag(qrow, 32 + 8 * hl);   // head dims 32..63

  float mOld = -1e30f, lsum = 0.f;
  v8f o0 = v8zero(), o1 = v8zero(), o2 = v8zero(), o3 = v8zero();

  const int ntiles = (qblk >> 5) + 2;      // causal: tiles up to block diagonal
  for (int t = 0; t < ntiles; ++t) {
    const int kvb = t * 32;

    // ---- stage K tile into LDS ----
#if HAVE_TDM
    if (wave == 0) {
      // 1-D TDM descriptor: 512 x 8B units = 32 rows x 128B, contiguous.
      size_t ga = (size_t)(Kw + ((size_t)b * TT + kvb) * HH);
      tdm_u4 g0 = { 1u,                                // count=1 (user D#)
                    ldsK,                              // lds_addr
                    (unsigned)ga,                      // global_addr[31:0]
                    ((unsigned)(ga >> 32) & 0x1FFFFFFu) | 0x80000000u };
                                                       // ga[56:32] | type=2
      tdm_i8 g1 = { 0x30000,                           // data_size=3 (8B)
                    (int)(512u << 16),                 // tensor_dim0 = 512
                    0,
                    (int)(512u << 16),                 // tile_dim0 = 512
                    0,
                    512,                               // tensor_dim0_stride
                    0, 0 };
      tdm_i4 z4 = { 0, 0, 0, 0 };
      tdm_i8 z8 = { 0, 0, 0, 0, 0, 0, 0, 0 };
      __builtin_amdgcn_tensor_load_to_lds(g0, g1, z4, z4, z8, 0);
      __builtin_amdgcn_s_wait_tensorcnt(0);
    }
#else
    {
      const uint4* gk = (const uint4*)(Kw + ((size_t)b * TT + kvb) * HH);
      uint4* sk = (uint4*)&kT[0][0];
      #pragma unroll
      for (int i = 0; i < 2; ++i) sk[tid + 128 * i] = gk[tid + 128 * i];
    }
#endif

    // ---- stage V tile transposed into LDS (all 128 threads) ----
    #pragma unroll
    for (int i = 0; i < 2; ++i) {
      int c   = tid + 128 * i;           // 256 chunks of 8 bf16
      int key = c >> 3;
      int h8  = (c & 7) * 8;
      uint4 d = *(const uint4*)(Vw + ((size_t)b * TT + kvb + key) * HH + h8);
      const __bf16* e = (const __bf16*)&d;
      #pragma unroll
      for (int j = 0; j < 8; ++j) vT[h8 + j][key] = e[j];
    }
    __syncthreads();

    // ---- S^T (keys x queries), rows = keys so softmax is per-lane ----
    v8f s0 = v8zero(), s1 = v8zero();
    {
      v16bf aK0a = lds_frag(&kT[ln][0],            8 * hl);
      s0 = wmma_bf16(aK0a, bq0, s0);
      v16bf aK0b = lds_frag(&kT[ln][0],       32 + 8 * hl);
      s0 = wmma_bf16(aK0b, bq1, s0);
      v16bf aK1a = lds_frag(&kT[16 + ln][0],       8 * hl);
      s1 = wmma_bf16(aK1a, bq0, s1);
      v16bf aK1b = lds_frag(&kT[16 + ln][0],  32 + 8 * hl);
      s1 = wmma_bf16(aK1b, bq1, s1);
    }

    // ---- causal mask + online softmax (in-lane + one lane^16 exchange) ----
    float e0[8], e1[8];
    float tmax = -1e30f;
    #pragma unroll
    for (int r = 0; r < 8; ++r) {
      int k0 = kvb + 8 * hl + r;           // key index of s0[r]
      int k1 = kvb + 16 + 8 * hl + r;      // key index of s1[r]
      e0[r] = (k0 <= qg) ? s0[r] : -1e30f;
      e1[r] = (k1 <= qg) ? s1[r] : -1e30f;
      tmax = fmaxf(tmax, fmaxf(e0[r], e1[r]));
    }
    tmax = fmaxf(tmax, __shfl_xor(tmax, 16));
    float mNew = fmaxf(mOld, tmax);
    float corr = __expf(mOld - mNew);
    float rs = 0.f;
    #pragma unroll
    for (int r = 0; r < 8; ++r) {
      e0[r] = __expf(e0[r] - mNew);
      e1[r] = __expf(e1[r] - mNew);
      rs += e0[r] + e1[r];
    }
    rs += __shfl_xor(rs, 16);
    lsum = lsum * corr + rs;
    mOld = mNew;
    o0 *= corr; o1 *= corr; o2 *= corr; o3 *= corr;

    // ---- P^T (D layout) -> B fragment: lane^16 exchange + bf16 repack ----
    float kv16[16];
    #pragma unroll
    for (int r = 0; r < 8; ++r) {
      float x0 = __shfl_xor(e0[r], 16);
      float x1 = __shfl_xor(e1[r], 16);
      kv16[r]     = e0[r];                 // keys 0..7   | 8..15
      kv16[8 + r] = e1[r];                 // keys 16..23 | 24..31
    }
    Frag16 bp;
    #pragma unroll
    for (int i = 0; i < 16; ++i) bp.e[i] = f2bf(kv16[i]);

    // ---- O^T += V^T @ P^T over 64 head dims (4 M-tiles) ----
    v16bf a0 = lds_frag(&vT[ln][0],      8 * hl);
    o0 = wmma_bf16(a0, bp.v, o0);
    v16bf a1 = lds_frag(&vT[16 + ln][0], 8 * hl);
    o1 = wmma_bf16(a1, bp.v, o1);
    v16bf a2 = lds_frag(&vT[32 + ln][0], 8 * hl);
    o2 = wmma_bf16(a2, bp.v, o2);
    v16bf a3 = lds_frag(&vT[48 + ln][0], 8 * hl);
    o3 = wmma_bf16(a3, bp.v, o3);

    __syncthreads();
  }

  // ---- normalize and store O (lane owns one query row; heads contiguous) --
  float inv = 1.f / lsum;
  float* orow = out + ((size_t)b * TT + qg) * HH + 8 * hl;
  v8f oo[4] = {o0, o1, o2, o3};
  #pragma unroll
  for (int t4 = 0; t4 < 4; ++t4) {
    float4 lo = { oo[t4][0]*inv, oo[t4][1]*inv, oo[t4][2]*inv, oo[t4][3]*inv };
    float4 hi = { oo[t4][4]*inv, oo[t4][5]*inv, oo[t4][6]*inv, oo[t4][7]*inv };
    *(float4*)(orow + 16 * t4)     = lo;
    *(float4*)(orow + 16 * t4 + 4) = hi;
  }
}

extern "C" void kernel_launch(void* const* d_in, const int* in_sizes, int n_in,
                              void* d_out, int out_size, void* d_ws, size_t ws_size,
                              hipStream_t stream) {
  (void)in_sizes; (void)n_in; (void)out_size; (void)ws_size;
  const float* x  = (const float*)d_in[0];
  const float* Wq = (const float*)d_in[1];
  const float* Wk = (const float*)d_in[2];
  const float* Wv = (const float*)d_in[3];
  float* out = (float*)d_out;

  const size_t nBTH = (size_t)BB * TT * HH;   // 1M elems
  const size_t nW   = (size_t)CC * HH;        // 64K elems
  __bf16* Qw  = (__bf16*)d_ws;                // 3 x 2MB QKV (bf16)
  __bf16* Kw  = Qw + nBTH;
  __bf16* Vw  = Kw + nBTH;
  __bf16* WqT = Vw + nBTH;                    // 3 x 128KB W^T (bf16)
  __bf16* WkT = WqT + nW;
  __bf16* WvT = WkT + nW;

  wtrans<<<dim3(HH, 3), 256, 0, stream>>>(Wq, Wk, Wv, WqT, WkT, WvT);
  proj_qkv<<<dim3((BB * TT) / 16), 128, 0, stream>>>(x, WqT, WkT, WvT,
                                                     Qw, Kw, Vw);
  attn_fwd<<<dim3(TT / 64, BB), 128, 0, stream>>>(Qw, Kw, Vw, out);
}
